// SelfAttention_10462540333620
// MI455X (gfx1250) — compile-verified, matrix operand order checked by reading, not run on hardware
//
#include <hip/hip_runtime.h>
#include <stdint.h>

// ---------------------------------------------------------------------------
// Causal self-attention for MI455X (gfx1250, wave32, WMMA).
//   B=4, S=2048, D=1024, fp32 in/out, bf16 WMMA with fp32 accumulation.
// Pipeline: cvt_x -> wtr (weight transpose) -> qkv_proj (fused 3-GEMM WMMA,
//           4-row-tile register blocking) -> attn (flash attention,
//           workgroup-cooperative, Q tile register-resident, WMMA).
// ---------------------------------------------------------------------------

#define D_DIM 1024
#define S_LEN 2048
#define B_N   4

typedef __attribute__((ext_vector_type(16))) __bf16 v16bf;
typedef __attribute__((ext_vector_type(8)))  float  v8f;

union Frag16 {
    v16bf v;
    uint4 q[2];
    unsigned short h[16];
};

// fp32 -> bf16 bits, round-to-nearest-even
__device__ __forceinline__ unsigned short f2bf(float f) {
    unsigned int u = __builtin_bit_cast(unsigned int, f);
    u += 0x7FFFu + ((u >> 16) & 1u);
    return (unsigned short)(u >> 16);
}

// Load a 16x32 bf16 A/B fragment for this lane.
// `base` points at this lane's first run: row_ptr + k0 + 8*(lane>>4).
// Elements 0..7  <- bf16[base + 0..7]   (K = k0 + 8*half + 0..7)
// Elements 8..15 <- bf16[base + 16..23] (K = k0 + 16 + 8*half + 0..7)
__device__ __forceinline__ v16bf load_frag(const unsigned short* base) {
    Frag16 f;
    f.q[0] = *(const uint4*)(base);
    f.q[1] = *(const uint4*)(base + 16);
    return f.v;
}

__device__ __forceinline__ v8f wmma_bf16(v16bf a, v16bf b, v8f c) {
    return __builtin_amdgcn_wmma_f32_16x16x32_bf16(
        /*neg_a=*/false, a, /*neg_b=*/false, b,
        /*c_mod=*/(short)0, c, /*reuse_a=*/false, /*reuse_b=*/false);
}

// ---------------------------------------------------------------------------
// Kernel 1: x fp32 -> bf16
// ---------------------------------------------------------------------------
__global__ void cvt_x(const float* __restrict__ x, unsigned short* __restrict__ xbf, int n) {
    int i = blockIdx.x * blockDim.x + threadIdx.x;
    if (i < n) xbf[i] = f2bf(x[i]);
}

// ---------------------------------------------------------------------------
// Kernel 2: weight transpose + convert.  WT[n*D + k] = bf16(W[k*D + n])
// blockIdx.y selects Q/K/V weight.
// ---------------------------------------------------------------------------
__global__ void wtr(const float* __restrict__ Qw, const float* __restrict__ Kw,
                    const float* __restrict__ Vw,
                    unsigned short* __restrict__ QwT, unsigned short* __restrict__ KwT,
                    unsigned short* __restrict__ VwT) {
    const float* src = (blockIdx.y == 0) ? Qw : (blockIdx.y == 1) ? Kw : Vw;
    unsigned short* dst = (blockIdx.y == 0) ? QwT : (blockIdx.y == 1) ? KwT : VwT;
    int i = blockIdx.x * blockDim.x + threadIdx.x;     // over D*D
    int k = i / D_DIM;
    int n = i % D_DIM;                                 // coalesced read along n
    dst[(size_t)n * D_DIM + k] = f2bf(src[(size_t)k * D_DIM + n]);
}

// ---------------------------------------------------------------------------
// Kernel 3: fused QKV projection with 4-row-tile register blocking.
// One wave per 64x16 output tile of EACH of Q,K,V: per 32-wide k-step it
// loads 4 x-A-fragments + 3 weight-B-fragments (14 b128 loads) and issues
// 12 WMMAs (~28 FLOP/byte from cache).  Q,K stored row-major bf16 [B*S][D];
// V stored transposed bf16 [B][D][S] so attention's P*V B-frags load
// contiguously.
// ---------------------------------------------------------------------------
#define QKV_ROWS 4   // 16-row tiles per wave

__global__ __launch_bounds__(128) void qkv_proj(
    const unsigned short* __restrict__ xbf,
    const unsigned short* __restrict__ QwT, const unsigned short* __restrict__ KwT,
    const unsigned short* __restrict__ VwT,
    unsigned short* __restrict__ Qbf, unsigned short* __restrict__ Kbf,
    unsigned short* __restrict__ VT) {

    const int wavesPerBlock = blockDim.x >> 5;
    int wave = blockIdx.x * wavesPerBlock + (threadIdx.x >> 5);
    int lane = threadIdx.x & 31;
    int half = lane >> 4;
    int l16  = lane & 15;

    const int colTiles = D_DIM / 16;             // 64
    int rowGroup = wave / colTiles;              // [0, B*S/(16*QKV_ROWS))
    int colTile  = wave % colTiles;
    int rowBase  = rowGroup * (16 * QKV_ROWS);

    const unsigned short* arow = xbf + (size_t)(rowBase + l16) * D_DIM + 8 * half;
    const unsigned short* qcol = QwT + (size_t)(colTile * 16 + l16) * D_DIM + 8 * half;
    const unsigned short* kcol = KwT + (size_t)(colTile * 16 + l16) * D_DIM + 8 * half;
    const unsigned short* vcol = VwT + (size_t)(colTile * 16 + l16) * D_DIM + 8 * half;

    v8f accQ[QKV_ROWS], accK[QKV_ROWS], accV[QKV_ROWS];
    #pragma unroll
    for (int j = 0; j < QKV_ROWS; j++) { accQ[j] = (v8f){}; accK[j] = (v8f){}; accV[j] = (v8f){}; }

    for (int k0 = 0; k0 < D_DIM; k0 += 32) {
        v16bf a[QKV_ROWS];
        #pragma unroll
        for (int j = 0; j < QKV_ROWS; j++)
            a[j] = load_frag(arow + (size_t)j * 16 * D_DIM + k0);
        v16bf bq = load_frag(qcol + k0);
        v16bf bk = load_frag(kcol + k0);
        v16bf bv = load_frag(vcol + k0);
        #pragma unroll
        for (int j = 0; j < QKV_ROWS; j++) {
            accQ[j] = wmma_bf16(a[j], bq, accQ[j]);
            accK[j] = wmma_bf16(a[j], bk, accK[j]);
            accV[j] = wmma_bf16(a[j], bv, accV[j]);
        }
    }

    int col = colTile * 16 + l16;
    #pragma unroll
    for (int j = 0; j < QKV_ROWS; j++) {
        int rb = rowBase + j * 16;
        // Q, K: row-major bf16 stores (C layout: VGPR r -> row rb + r + 8*half)
        #pragma unroll
        for (int r = 0; r < 8; r++) {
            int row = rb + r + 8 * half;
            Qbf[(size_t)row * D_DIM + col] = f2bf(accQ[j][r]);
            Kbf[(size_t)row * D_DIM + col] = f2bf(accK[j][r]);
        }
        // V transposed: VT[(b*D + col)*S + s]; r -> consecutive s => one b128 store
        int b    = rb / S_LEN;
        int sInB = rb % S_LEN;
        union { uint4 q; unsigned short h[8]; } pv;
        #pragma unroll
        for (int r = 0; r < 8; r++) pv.h[r] = f2bf(accV[j][r]);
        *(uint4*)(VT + (size_t)(b * D_DIM + col) * S_LEN + sInB + 8 * half) = pv.q;
    }
}

// ---------------------------------------------------------------------------
// Kernel 4: flash attention with causal mask.
// Workgroup = 4 waves = one (batch, 16-query block).  Wave w owns d-slice
// [w*256, w*256+256).  The wave's Q slice (8 fragments, 64 VGPRs) is loaded
// ONCE and stays register-resident for all key blocks.  Per 32-key block:
// partial QK^T over the d-slice (16 WMMAs), LDS sum across waves, fp32
// online softmax in exp2 domain, P redistributed through LDS into A-frag
// layout, then 16 P*V WMMAs into the 16x256 fp32 accumulator.
// ---------------------------------------------------------------------------
__global__ __launch_bounds__(128) void attn(
    const unsigned short* __restrict__ Qbf, const unsigned short* __restrict__ Kbf,
    const unsigned short* __restrict__ VT, float* __restrict__ out) {

    __shared__ float lds_part[4][16][32];   // per-wave partial scores (8 KB)
    __shared__ float lds_P[16][32];         // exp'd probabilities (2 KB)

    int w    = threadIdx.x >> 5;
    int lane = threadIdx.x & 31;
    int half = lane >> 4;
    int l16  = lane & 15;

    int b  = blockIdx.x / (S_LEN / 16);
    int qb = blockIdx.x % (S_LEN / 16);
    size_t batchRow = (size_t)b * S_LEN;

    int d0w = w * 256;

    // Preload this wave's Q slice: 16 rows x 256 d, register-resident.
    const unsigned short* Qrow = Qbf + (batchRow + qb * 16 + l16) * D_DIM + 8 * half;
    v16bf qf[8];
    #pragma unroll
    for (int ds8 = 0; ds8 < 8; ds8++)
        qf[ds8] = load_frag(Qrow + d0w + ds8 * 32);

    v8f acc[16];
    #pragma unroll
    for (int t = 0; t < 16; t++) acc[t] = (v8f){};
    float m[8], l[8];
    #pragma unroll
    for (int r = 0; r < 8; r++) { m[r] = -1e30f; l[r] = 0.0f; }

    const float SC = 1.4426950408889634f / 32.0f;  // log2(e)/sqrt(D)
    int nkb = (qb * 16 + 15) / 32 + 1;             // causal key-block count

    for (int kb = 0; kb < nkb; kb++) {
        int k0key = kb * 32;

        // ---- partial scores over this wave's d-slice ----
        v8f s0 = {}; v8f s1 = {};
        const unsigned short* Krow0 = Kbf + (batchRow + k0key + l16) * D_DIM + 8 * half;
        const unsigned short* Krow1 = Krow0 + (size_t)16 * D_DIM;
        #pragma unroll
        for (int ds8 = 0; ds8 < 8; ds8++) {
            v16bf bk0 = load_frag(Krow0 + d0w + ds8 * 32);
            v16bf bk1 = load_frag(Krow1 + d0w + ds8 * 32);
            s0 = wmma_bf16(qf[ds8], bk0, s0);
            s1 = wmma_bf16(qf[ds8], bk1, s1);
        }
        #pragma unroll
        for (int r = 0; r < 8; r++) {
            lds_part[w][r + 8 * half][l16]      = s0[r];
            lds_part[w][r + 8 * half][l16 + 16] = s1[r];
        }
        __syncthreads();

        // ---- sum partials, online softmax (per row M = r + 8*half) ----
        #pragma unroll
        for (int r = 0; r < 8; r++) {
            int M = r + 8 * half;
            float v0 = lds_part[0][M][l16] + lds_part[1][M][l16] +
                       lds_part[2][M][l16] + lds_part[3][M][l16];
            float v1 = lds_part[0][M][l16 + 16] + lds_part[1][M][l16 + 16] +
                       lds_part[2][M][l16 + 16] + lds_part[3][M][l16 + 16];
            int qrow = qb * 16 + M;
            float sl0 = (k0key + l16      <= qrow) ? v0 * SC : -1e30f;
            float sl1 = (k0key + l16 + 16 <= qrow) ? v1 * SC : -1e30f;

            float mx = fmaxf(sl0, sl1);
            mx = fmaxf(mx, __shfl_xor(mx, 1));
            mx = fmaxf(mx, __shfl_xor(mx, 2));
            mx = fmaxf(mx, __shfl_xor(mx, 4));
            mx = fmaxf(mx, __shfl_xor(mx, 8));

            float mn    = fmaxf(m[r], mx);
            float scale = exp2f(m[r] - mn);
            float p0    = exp2f(sl0 - mn);
            float p1    = exp2f(sl1 - mn);
            float rs = p0 + p1;
            rs += __shfl_xor(rs, 1);
            rs += __shfl_xor(rs, 2);
            rs += __shfl_xor(rs, 4);
            rs += __shfl_xor(rs, 8);

            l[r] = l[r] * scale + rs;
            m[r] = mn;
            #pragma unroll
            for (int t = 0; t < 16; t++) acc[t][r] *= scale;

            if (w == 0) { lds_P[M][l16] = p0; lds_P[M][l16 + 16] = p1; }
        }
        __syncthreads();

        // ---- build P A-fragment from LDS (A layout: M = lane%16) ----
        Frag16 pf;
        #pragma unroll
        for (int e = 0; e < 16; e++) {
            int k = (e < 8) ? (e + 8 * half) : (16 + (e - 8) + 8 * half);
            pf.h[e] = f2bf(lds_P[l16][k]);
        }

        // ---- P * V into accumulators (V^T gives contiguous K runs) ----
        const unsigned short* vbase =
            VT + (size_t)(b * D_DIM + d0w + l16) * S_LEN + k0key + 8 * half;
        #pragma unroll
        for (int t = 0; t < 16; t++) {
            v16bf bv = load_frag(vbase + (size_t)t * 16 * S_LEN);
            acc[t] = wmma_bf16(pf.v, bv, acc[t]);
        }
    }

    // ---- epilogue: normalize and store fp32 ----
    #pragma unroll
    for (int t = 0; t < 16; t++) {
        #pragma unroll
        for (int r = 0; r < 8; r++) {
            int row = qb * 16 + r + 8 * half;
            out[(batchRow + row) * D_DIM + d0w + t * 16 + l16] = acc[t][r] / l[r];
        }
    }
}

// ---------------------------------------------------------------------------
extern "C" void kernel_launch(void* const* d_in, const int* in_sizes, int n_in,
                              void* d_out, int out_size, void* d_ws, size_t ws_size,
                              hipStream_t stream) {
    const float* x  = (const float*)d_in[0];
    const float* Qw = (const float*)d_in[1];
    const float* Kw = (const float*)d_in[2];
    const float* Vw = (const float*)d_in[3];
    float* out = (float*)d_out;

    const size_t MB = 1024 * 1024;
    char* ws = (char*)d_ws;
    unsigned short* xbf = (unsigned short*)(ws + 0 * MB);   // 16 MB
    unsigned short* Qbf = (unsigned short*)(ws + 16 * MB);  // 16 MB
    unsigned short* Kbf = (unsigned short*)(ws + 32 * MB);  // 16 MB
    unsigned short* VT  = (unsigned short*)(ws + 48 * MB);  // 16 MB
    unsigned short* QwT = (unsigned short*)(ws + 64 * MB);  // 2 MB
    unsigned short* KwT = (unsigned short*)(ws + 66 * MB);  // 2 MB
    unsigned short* VwT = (unsigned short*)(ws + 68 * MB);  // 2 MB

    const int nX = B_N * S_LEN * D_DIM;                     // 8,388,608
    cvt_x<<<nX / 256, 256, 0, stream>>>(x, xbf, nX);
    wtr<<<dim3(D_DIM * D_DIM / 256, 3), 256, 0, stream>>>(Qw, Kw, Vw, QwT, KwT, VwT);

    // one wave per 64x16 tile group: (B*S/64) * (D/16) = 128*64 = 8192 waves
    qkv_proj<<<8192 / 4, 128, 0, stream>>>(xbf, QwT, KwT, VwT, Qbf, Kbf, VT);

    // one workgroup (4 waves) per (batch, 16-query block): 4 * 128 = 512
    attn<<<B_N * (S_LEN / 16), 128, 0, stream>>>(Qbf, Kbf, VT, out);
}
